// regression_Transformer_33062658245008
// MI455X (gfx1250) — matrix operand order repaired and checked by reading, not compiled, not run on hardware
//
#include <hip/hip_runtime.h>

// ---------------- problem constants ----------------
constexpr int Bb  = 32;
constexpr int Ss  = 512;
constexpr int Dd  = 256;
constexpr int Hh  = 8;
constexpr int Ll  = 6;
constexpr int FINc = 7;
constexpr int FFc = 1024;
constexpr int BS  = Bb * Ss;          // 16384 rows

typedef __attribute__((ext_vector_type(16))) _Float16 v16h;
typedef __attribute__((ext_vector_type(8)))  _Float16 v8h;
typedef __attribute__((ext_vector_type(8)))  float    v8f;
typedef __attribute__((ext_vector_type(4)))  unsigned int u32x4;
typedef __attribute__((ext_vector_type(8)))  int      i32x8;
typedef __attribute__((ext_vector_type(4)))  int      i32x4;

// Build a 16x32 f16 A-operand fragment: lane holds row (lane&15),
// K chunk [2*hi..2*hi+15] (hi = 8*(lane>>4)).
__device__ __forceinline__ v16h load_afrag(const _Float16* row, int k0, int hi) {
  v8h lo = *(const v8h*)(row + k0 + 2 * hi);
  v8h hi8 = *(const v8h*)(row + k0 + 2 * hi + 8);
  v16h f;
#pragma unroll
  for (int i = 0; i < 8; ++i) { f[i] = lo[i]; f[8 + i] = hi8[i]; }
  return f;
}

// ---------------- TDM: 2-D f16 tile (rows x cols) global -> LDS ----------------
// D# built per CDNA5 ISA ch.8: group0 {count=1, lds_addr, global_addr, type=2},
// group1 {data_size=2B, tensor_dim0/1, tile_dim0/1, tensor_dim0_stride}.
// This toolchain exposes the 6-arg builtin: (g0, g1, g2, g3, extra, cpol).
__device__ __forceinline__ void tdm_load_2d(unsigned lds_off, const void* gptr,
                                            unsigned rows, unsigned cols,
                                            unsigned stride_elems) {
  unsigned long long ga = (unsigned long long)(uintptr_t)gptr;
  u32x4 g0;
  g0[0] = 1u;                                              // count=1 (valid), user mode
  g0[1] = lds_off;                                         // lds_addr (bytes)
  g0[2] = (unsigned)(ga & 0xffffffffu);                    // global_addr[31:0]
  g0[3] = (unsigned)((ga >> 32) & 0x01ffffffu) | (2u << 30); // addr[56:32] | type=2
  i32x8 g1;
  g1[0] = (int)(1u << 16);                                 // data_size=1 -> 2 bytes
  g1[1] = (int)((cols & 0xffffu) << 16);                   // tensor_dim0[15:0]
  g1[2] = (int)(((cols >> 16) & 0xffffu) | ((rows & 0xffffu) << 16)); // td0 hi | td1 lo
  g1[3] = (int)(((rows >> 16) & 0xffffu) | ((cols & 0xffffu) << 16)); // td1 hi | tile_dim0
  g1[4] = (int)(rows & 0xffffu);                           // tile_dim1 | tile_dim2=0
  g1[5] = (int)stride_elems;                               // tensor_dim0_stride[31:0]
  g1[6] = 0;
  g1[7] = 0;
  i32x4 z4 = {0, 0, 0, 0};
  i32x8 z8 = {0, 0, 0, 0, 0, 0, 0, 0};
  __builtin_amdgcn_tensor_load_to_lds(g0, g1, z4, z4, z8, 0);
}

// ---------------- weight convert + transpose (f32 [K,N] -> f16 [N,K]) ----------------
__global__ void convT_k(const float* __restrict__ W, _Float16* __restrict__ Wt,
                        int K, int N) {
  long long base = (long long)blockIdx.z * K * N;
  int idx = blockIdx.x * blockDim.x + threadIdx.x;
  if (idx >= K * N) return;
  int n = idx % N, k = idx / N;
  Wt[base + (long long)n * K + k] = (_Float16)W[base + idx];
}

// ---------------- embedding: h = x @ W_in + b_in + pos ----------------
__global__ void embed_k(const float* __restrict__ x, const float* __restrict__ Win,
                        const float* __restrict__ bin, const float* __restrict__ pos,
                        float* __restrict__ h, _Float16* __restrict__ hb) {
  int idx = blockIdx.x * blockDim.x + threadIdx.x;
  if (idx >= BS * Dd) return;
  int d  = idx & (Dd - 1);
  int bs = idx >> 8;
  int s  = bs & (Ss - 1);
  float acc = bin[d] + pos[(size_t)s * Dd + d];
  const float* xr = x + (size_t)bs * FINc;
#pragma unroll
  for (int f = 0; f < FINc; ++f) acc += xr[f] * Win[f * Dd + d];
  h[idx] = acc;
  hb[idx] = (_Float16)acc;
}

// ---------------- WMMA GEMM: C[M,N] = A[M,K](f16) @ Wt[N,K](f16) + bias ----------------
// One wave computes a 16x64 strip (4 accumulators share one A fragment).
__global__ __launch_bounds__(256) void gemm16(
    const _Float16* __restrict__ A, const _Float16* __restrict__ Wt,
    const float* __restrict__ bias, const float* __restrict__ residual,
    float* __restrict__ Cf, _Float16* __restrict__ Cb,
    int M, int N, int K,
    long long wBatch, long long cBatch, long long biasBatch,
    int relu, int storeVT) {
  int wave = threadIdx.x >> 5;
  int lane = threadIdx.x & 31;
  int hlf  = lane >> 4;
  int lrow = lane & 15;
  int hi   = hlf * 8;

  int tid  = blockIdx.x * 8 + wave;
  int nGrp = N >> 6;                 // 64-col groups
  int m0 = (tid / nGrp) << 4;
  int n0 = (tid % nGrp) << 6;
  if (m0 >= M) return;

  int z = blockIdx.z;
  const _Float16* arow  = A + (size_t)(m0 + lrow) * K;
  const _Float16* wbase = Wt + (long long)z * wBatch;

  v8f acc[4];
#pragma unroll
  for (int t = 0; t < 4; ++t) {
    float bv = 0.f;
    if (bias) bv = bias[(long long)z * biasBatch + n0 + t * 16 + lrow];
#pragma unroll
    for (int i = 0; i < 8; ++i) acc[t][i] = bv;
  }

  for (int k0 = 0; k0 < K; k0 += 32) {
    v16h af = load_afrag(arow, k0, hi);
#pragma unroll
    for (int t = 0; t < 4; ++t) {
      const _Float16* wrow = wbase + (size_t)(n0 + t * 16 + lrow) * K;
      v16h bf = *(const v16h*)(wrow + k0 + hlf * 16);
      acc[t] = __builtin_amdgcn_wmma_f32_16x16x32_f16(false, af, false, bf,
                                                      (short)0, acc[t], false, false);
    }
  }

#pragma unroll
  for (int t = 0; t < 4; ++t) {
#pragma unroll
    for (int r = 0; r < 8; ++r) {
      int row = m0 + r + hlf * 8;            // C/D layout: M = r + 8*(lane>>4)
      int col = n0 + t * 16 + lrow;          //            N = lane&15
      float v = acc[t][r];
      if (residual) v += residual[(size_t)row * N + col];
      if (Cf) Cf[(size_t)row * N + col] = v;
      if (relu) v = v > 0.f ? v : 0.f;
      if (Cb) {
        if (storeVT) {                       // store V transposed: [H][B][D][S]
          int bb = row >> 9;
          int ss = row & (Ss - 1);
          Cb[(long long)z * cBatch + ((size_t)bb * N + col) * Ss + ss] = (_Float16)v;
        } else {
          Cb[(long long)z * cBatch + (size_t)row * N + col] = (_Float16)v;
        }
      }
    }
  }
}

// ---------------- attention: one wave per (b, h, 16-row q tile) ----------------
// K and V^T tiles staged through LDS with double-buffered TDM async loads.
constexpr int KCH = 64;                  // K rows per TDM chunk
constexpr int NCH = Ss / KCH;            // 8 chunks
constexpr unsigned KST_OFF = 0;                              // 2*64*256*2 = 64KB
constexpr unsigned VST_OFF = KST_OFF + 2 * KCH * Dd * 2;     // 2*16*512*2 = 32KB
constexpr unsigned SC_OFF  = VST_OFF + 2 * 16 * Ss * 2;      // 16*512*4  = 32KB
constexpr unsigned PB_OFF  = SC_OFF + 16 * Ss * 4;           // 16*512*2  = 16KB
constexpr unsigned SMEM_BYTES = PB_OFF + 16 * Ss * 2;        // 144KB total

__global__ __launch_bounds__(32) void attention_k(
    const _Float16* __restrict__ q, const _Float16* __restrict__ kmat,
    const _Float16* __restrict__ vt, const int* __restrict__ lens,
    _Float16* __restrict__ obuf) {
  __shared__ __align__(128) unsigned char smem[SMEM_BYTES];
  _Float16* kst = (_Float16*)(smem + KST_OFF);
  _Float16* vst = (_Float16*)(smem + VST_OFF);
  float*    sc  = (float*)(smem + SC_OFF);
  _Float16* pb  = (_Float16*)(smem + PB_OFF);

  int lane = threadIdx.x;
  int hlf = lane >> 4, lrow = lane & 15, hi = hlf * 8;
  int qt = blockIdx.x, hH = blockIdx.y, b = blockIdx.z;
  int len = lens[b];
  size_t hb_off = (size_t)hH * Bb + b;
  const _Float16* qb = q    + hb_off * Ss * Dd;
  const _Float16* kb = kmat + hb_off * Ss * Dd;
  const _Float16* vb = vt   + hb_off * (size_t)Dd * Ss;
  int m0 = qt * 16;
  const _Float16* arow = qb + (size_t)(m0 + lrow) * Dd;
  const float scale = 0.0625f;  // 1/sqrt(256)

  // ---- phase 1: scores = Q @ K^T, K chunks DMA'd into LDS (double buffered) ----
  tdm_load_2d(KST_OFF, kb, KCH, Dd, Dd);
  for (int ch = 0; ch < NCH; ++ch) {
    if (ch + 1 < NCH) {
      unsigned dst = KST_OFF + (unsigned)(((ch + 1) & 1) * KCH * Dd * 2);
      tdm_load_2d(dst, kb + (size_t)(ch + 1) * KCH * Dd, KCH, Dd, Dd);
      __builtin_amdgcn_s_wait_tensorcnt(1);   // current chunk resident
    } else {
      __builtin_amdgcn_s_wait_tensorcnt(0);
    }
    const _Float16* kbase = kst + ((ch & 1) ? KCH * Dd : 0);

    v8f acc[4];
#pragma unroll
    for (int t = 0; t < 4; ++t)
#pragma unroll
      for (int i = 0; i < 8; ++i) acc[t][i] = 0.f;

    for (int k0 = 0; k0 < Dd; k0 += 32) {
      v16h af = load_afrag(arow, k0, hi);
#pragma unroll
      for (int t = 0; t < 4; ++t) {
        v16h bf = *(const v16h*)(kbase + (size_t)(t * 16 + lrow) * Dd + k0 + hlf * 16);
        acc[t] = __builtin_amdgcn_wmma_f32_16x16x32_f16(false, af, false, bf,
                                                        (short)0, acc[t], false, false);
      }
    }
#pragma unroll
    for (int t = 0; t < 4; ++t) {
#pragma unroll
      for (int r = 0; r < 8; ++r) {
        int row = r + hlf * 8;
        int col = ch * KCH + t * 16 + lrow;
        int qrow = m0 + row;
        sc[row * Ss + col] = (col < len && qrow < len) ? acc[t][r] * scale : -1e9f;
      }
    }
  }

  // ---- softmax over 512 cols; two lanes per row, combine across lane^16 ----
  {
    int row = lrow;
    int c0 = hlf * 256;
    float mx = -3.0e38f;
    for (int c = 0; c < 256; ++c) mx = fmaxf(mx, sc[row * Ss + c0 + c]);
    mx = fmaxf(mx, __shfl_xor(mx, 16));
    float sum = 0.f;
    for (int c = 0; c < 256; ++c) {
      float e = __expf(sc[row * Ss + c0 + c] - mx);
      sum += e;
      sc[row * Ss + c0 + c] = e;
    }
    sum += __shfl_xor(sum, 16);
    float inv = 1.f / sum;
    for (int c = 0; c < 256; ++c)
      pb[row * Ss + c0 + c] = (_Float16)(sc[row * Ss + c0 + c] * inv);
  }
  // single wave: LDS ops are in-order per wave, no barrier needed

  // ---- phase 2: O = P @ V, V^T 16-row tiles DMA'd into LDS (double buffered) ----
  tdm_load_2d(VST_OFF, vb, 16, Ss, Ss);
  for (int ntt = 0; ntt < Dd / 16; ++ntt) {
    int nt = ntt * 16;
    if (ntt + 1 < Dd / 16) {
      unsigned dst = VST_OFF + (unsigned)(((ntt + 1) & 1) * 16 * Ss * 2);
      tdm_load_2d(dst, vb + (size_t)(nt + 16) * Ss, 16, Ss, Ss);
      __builtin_amdgcn_s_wait_tensorcnt(1);
    } else {
      __builtin_amdgcn_s_wait_tensorcnt(0);
    }
    const _Float16* vbase = vst + ((ntt & 1) ? 16 * Ss : 0);

    v8f acc;
#pragma unroll
    for (int i = 0; i < 8; ++i) acc[i] = 0.f;
    for (int k0 = 0; k0 < Ss; k0 += 32) {
      v16h af = load_afrag(pb + (size_t)lrow * Ss, k0, hi);
      v16h bf = *(const v16h*)(vbase + (size_t)lrow * Ss + k0 + hlf * 16);
      acc = __builtin_amdgcn_wmma_f32_16x16x32_f16(false, af, false, bf,
                                                   (short)0, acc, false, false);
    }
#pragma unroll
    for (int r = 0; r < 8; ++r) {
      int qrow = m0 + r + hlf * 8;
      int col = nt + lrow;
      obuf[((size_t)(b * Ss + qrow)) * (Hh * Dd) + hH * Dd + col] = (_Float16)acc[r];
    }
  }
}

// ---------------- layernorm: one wave per row of 256 ----------------
__global__ __launch_bounds__(256) void layernorm_k(
    const float* __restrict__ t, const float* __restrict__ g,
    const float* __restrict__ be, float* __restrict__ outf,
    _Float16* __restrict__ outh) {
  int wave = threadIdx.x >> 5, lane = threadIdx.x & 31;
  int row = blockIdx.x * 8 + wave;
  const float* tr = t + (size_t)row * Dd;
  float vals[8];
  float s = 0.f;
#pragma unroll
  for (int i = 0; i < 8; ++i) { vals[i] = tr[lane + i * 32]; s += vals[i]; }
#pragma unroll
  for (int m = 16; m >= 1; m >>= 1) s += __shfl_xor(s, m);
  float mu = s * (1.0f / Dd);
  float var = 0.f;
#pragma unroll
  for (int i = 0; i < 8; ++i) { float d = vals[i] - mu; var += d * d; }
#pragma unroll
  for (int m = 16; m >= 1; m >>= 1) var += __shfl_xor(var, m);
  float inv = rsqrtf(var * (1.0f / Dd) + 1e-5f);
#pragma unroll
  for (int i = 0; i < 8; ++i) {
    int di = lane + i * 32;
    float o = (vals[i] - mu) * inv * g[di] + be[di];
    outf[(size_t)row * Dd + di] = o;
    outh[(size_t)row * Dd + di] = (_Float16)o;
  }
}

// ---------------- masked mean pool + regression head ----------------
__global__ __launch_bounds__(256) void pool_head_k(
    const float* __restrict__ h, const int* __restrict__ lens,
    const float* __restrict__ Wr, const float* __restrict__ br,
    float* __restrict__ out) {
  __shared__ float red[256];
  int b = blockIdx.x, d = threadIdx.x;
  int len = lens[b];
  float acc = 0.f;
  for (int s = 0; s < len; ++s) acc += h[((size_t)(b * Ss + s)) * Dd + d];
  acc = (acc / (float)len) * Wr[d];
  red[d] = acc;
  __syncthreads();
  for (int o = 128; o >= 1; o >>= 1) {
    if (d < o) red[d] += red[d + o];
    __syncthreads();
  }
  if (d == 0) out[b] = red[0] + br[0];
}

// ---------------- host orchestration ----------------
extern "C" void kernel_launch(void* const* d_in, const int* in_sizes, int n_in,
                              void* d_out, int out_size, void* d_ws, size_t ws_size,
                              hipStream_t stream) {
  (void)in_sizes; (void)n_in; (void)out_size; (void)ws_size;
  const float* x   = (const float*)d_in[0];
  const int*   lens= (const int*)d_in[1];
  const float* Win = (const float*)d_in[2];
  const float* bin = (const float*)d_in[3];
  const float* pos = (const float*)d_in[4];
  const float* Wq  = (const float*)d_in[5];
  const float* bq  = (const float*)d_in[6];
  const float* Wk  = (const float*)d_in[7];
  const float* bk  = (const float*)d_in[8];
  const float* Wv  = (const float*)d_in[9];
  const float* bv  = (const float*)d_in[10];
  const float* Wo  = (const float*)d_in[11];
  const float* bo  = (const float*)d_in[12];
  const float* W1  = (const float*)d_in[13];
  const float* b1  = (const float*)d_in[14];
  const float* W2  = (const float*)d_in[15];
  const float* b2  = (const float*)d_in[16];
  const float* g1  = (const float*)d_in[17];
  const float* be1 = (const float*)d_in[18];
  const float* g2  = (const float*)d_in[19];
  const float* be2 = (const float*)d_in[20];
  const float* Wr  = (const float*)d_in[21];
  const float* br  = (const float*)d_in[22];
  float* out = (float*)d_out;

  char* ws = (char*)d_ws;
  size_t off = 0;
  auto alloc = [&](size_t bytes) -> void* {
    void* p = ws + off;
    off += (bytes + 255) & ~(size_t)255;
    return p;
  };
  float*    h    = (float*)   alloc((size_t)BS * Dd * 4);
  _Float16* hb   = (_Float16*)alloc((size_t)BS * Dd * 2);
  _Float16* qb   = (_Float16*)alloc((size_t)Hh * BS * Dd * 2);
  _Float16* kbuf = (_Float16*)alloc((size_t)Hh * BS * Dd * 2);
  _Float16* vtb  = (_Float16*)alloc((size_t)Hh * BS * Dd * 2);
  _Float16* ob   = (_Float16*)alloc((size_t)BS * Hh * Dd * 2);
  float*    t1   = (float*)   alloc((size_t)BS * Dd * 4);
  float*    h1   = (float*)   alloc((size_t)BS * Dd * 4);
  _Float16* h1b  = (_Float16*)alloc((size_t)BS * Dd * 2);
  _Float16* ffb  = (_Float16*)alloc((size_t)BS * FFc * 2);
  _Float16* WqT  = (_Float16*)alloc((size_t)Ll * Hh * Dd * Dd * 2);
  _Float16* WkT  = (_Float16*)alloc((size_t)Ll * Hh * Dd * Dd * 2);
  _Float16* WvT  = (_Float16*)alloc((size_t)Ll * Hh * Dd * Dd * 2);
  _Float16* WoT  = (_Float16*)alloc((size_t)Ll * Dd * Hh * Dd * 2);
  _Float16* W1T  = (_Float16*)alloc((size_t)Ll * FFc * Dd * 2);
  _Float16* W2T  = (_Float16*)alloc((size_t)Ll * Dd * FFc * 2);

  // one-time weight convert+transpose (all weights fit in 192MB L2 afterwards)
  {
    int e = Dd * Dd;
    dim3 g((e + 255) / 256, 1, Ll * Hh);
    convT_k<<<g, 256, 0, stream>>>(Wq, WqT, Dd, Dd);
    convT_k<<<g, 256, 0, stream>>>(Wk, WkT, Dd, Dd);
    convT_k<<<g, 256, 0, stream>>>(Wv, WvT, Dd, Dd);
  }
  { int e = Hh * Dd * Dd; dim3 g((e + 255) / 256, 1, Ll);
    convT_k<<<g, 256, 0, stream>>>(Wo, WoT, Hh * Dd, Dd); }
  { int e = Dd * FFc;     dim3 g((e + 255) / 256, 1, Ll);
    convT_k<<<g, 256, 0, stream>>>(W1, W1T, Dd, FFc); }
  { int e = FFc * Dd;     dim3 g((e + 255) / 256, 1, Ll);
    convT_k<<<g, 256, 0, stream>>>(W2, W2T, FFc, Dd); }

  embed_k<<<(BS * Dd + 255) / 256, 256, 0, stream>>>(x, Win, bin, pos, h, hb);

  const int grpD  = (BS / 16) * (Dd / 64);   // 4096 strips -> 512 blocks
  const int grpFF = (BS / 16) * (FFc / 64);  // 16384 strips -> 2048 blocks

  for (int l = 0; l < Ll; ++l) {
    dim3 gq(grpD / 8, 1, Hh);
    // Q, K: normal layout; V: stored transposed [H][B][D][S]
    gemm16<<<gq, 256, 0, stream>>>(hb, WqT + (size_t)l * Hh * Dd * Dd,
        bq + (size_t)l * Hh * Dd, nullptr, nullptr, qb,
        BS, Dd, Dd, (long long)Dd * Dd, (long long)BS * Dd, Dd, 0, 0);
    gemm16<<<gq, 256, 0, stream>>>(hb, WkT + (size_t)l * Hh * Dd * Dd,
        bk + (size_t)l * Hh * Dd, nullptr, nullptr, kbuf,
        BS, Dd, Dd, (long long)Dd * Dd, (long long)BS * Dd, Dd, 0, 0);
    gemm16<<<gq, 256, 0, stream>>>(hb, WvT + (size_t)l * Hh * Dd * Dd,
        bv + (size_t)l * Hh * Dd, nullptr, nullptr, vtb,
        BS, Dd, Dd, (long long)Dd * Dd, (long long)Bb * Dd * Ss, Dd, 0, 1);

    dim3 ga(Ss / 16, Hh, Bb);
    attention_k<<<ga, 32, 0, stream>>>(qb, kbuf, vtb, lens, ob);

    // t1 = h + ob @ Wo + bo
    gemm16<<<dim3(grpD / 8, 1, 1), 256, 0, stream>>>(ob,
        WoT + (size_t)l * Dd * Hh * Dd, bo + (size_t)l * Dd,
        h, t1, nullptr, BS, Dd, Hh * Dd, 0, 0, 0, 0, 0);
    layernorm_k<<<BS / 8, 256, 0, stream>>>(t1, g1 + l * Dd, be1 + l * Dd, h1, h1b);

    // ffb = relu(h1 @ W1 + b1)
    gemm16<<<dim3(grpFF / 8, 1, 1), 256, 0, stream>>>(h1b,
        W1T + (size_t)l * FFc * Dd, b1 + (size_t)l * FFc,
        nullptr, nullptr, ffb, BS, FFc, Dd, 0, 0, 0, 1, 0);
    // t1 = h1 + ffb @ W2 + b2
    gemm16<<<dim3(grpD / 8, 1, 1), 256, 0, stream>>>(ffb,
        W2T + (size_t)l * Dd * FFc, b2 + (size_t)l * Dd,
        h1, t1, nullptr, BS, Dd, FFc, 0, 0, 0, 0, 0);
    layernorm_k<<<BS / 8, 256, 0, stream>>>(t1, g2 + l * Dd, be2 + l * Dd, h, hb);
  }

  pool_head_k<<<Bb, 256, 0, stream>>>(h, lens, Wr, br, out);
}
